// DeepARV1_78159814853285
// MI455X (gfx1250) — matrix operand order repaired and checked by reading, not compiled, not run on hardware
//
#include <hip/hip_runtime.h>
#include <hip/hip_bf16.h>
#include <stdint.h>

#define HID     1024
#define BATCH   1024
#define LAYERS  3
#define GATES   (4 * HID)   // 4096
#define IN_DIM  69
#define IN_PAD  96
#define EMB_DIM 64

typedef __attribute__((ext_vector_type(16))) __bf16 v16bf;
typedef __attribute__((ext_vector_type(8)))  __bf16 v8bf;
typedef __attribute__((ext_vector_type(8)))  float  v8f;

// Pack two fp32 into packed bf16 pair (round half-up) with one v_perm_b32.
__device__ __forceinline__ unsigned pack2(float a, float b) {
    unsigned ua = __float_as_uint(a) + 0x8000u;
    unsigned ub = __float_as_uint(b) + 0x8000u;
    // result bytes: {ub.b3, ub.b2, ua.b3, ua.b2} -> low16 = bf16(a), high16 = bf16(b)
    return __builtin_amdgcn_perm(ub, ua, 0x07060302u);
}

// ---------------------------------------------------------------------------
// Kernel 1a: build padded input row [x(5) | emb[idx](64) | zeros(27)]
// ---------------------------------------------------------------------------
__global__ void build_input(const float* __restrict__ x, const int* __restrict__ idx,
                            const float* __restrict__ emb, float* __restrict__ inp0) {
    int i = blockIdx.x * blockDim.x + threadIdx.x;
    if (i >= BATCH * IN_PAD) return;
    int b = i / IN_PAD, c = i - b * IN_PAD;
    float v = 0.f;
    if (c < 5)            v = x[b * 5 + c];
    else if (c < IN_DIM)  v = emb[(size_t)idx[b] * EMB_DIM + (c - 5)];
    inp0[i] = v;
}

// ---------------------------------------------------------------------------
// Kernel 1b: zero-pad w_ih0 [4096,69] -> [4096,96] so GEMM staging is
// branch-free (aligned rows, K multiple of 32).
// ---------------------------------------------------------------------------
__global__ void pad_w0(const float* __restrict__ w, float* __restrict__ wp) {
    int i = blockIdx.x * blockDim.x + threadIdx.x;
    if (i >= GATES * IN_PAD) return;
    int r = i / IN_PAD, c = i - r * IN_PAD;
    wp[i] = (c < IN_DIM) ? w[r * IN_DIM + c] : 0.f;
}

// ---------------------------------------------------------------------------
// Kernel 2: gates[1024,4096] = A0 @ W0^T + A1 @ W1^T   (bias deferred)
// K-concatenated GEMM, bf16 WMMA 16x16x32, fp32 accumulate.
// Block: 256 threads (8 waves), 128x128 tile, K-chunk 32, double-buffered LDS.
// Preconditions: all row strides 16B-aligned, chunk counts n0,n1 exact.
// ---------------------------------------------------------------------------
__launch_bounds__(256)
__global__ void gates_gemm(const float* __restrict__ A0, int lda0, int n0,
                           const float* __restrict__ W0, int ldw0,
                           const float* __restrict__ A1,
                           const float* __restrict__ W1, int n1,
                           float* __restrict__ gates) {
    __shared__ __align__(16) __bf16 As[2][128][40];   // stride 40: conflict-free b128
    __shared__ __align__(16) __bf16 Bs[2][128][40];

    const int t      = threadIdx.x;
    const int lane   = t & 31;
    const int wave   = t >> 5;
    const int m16    = lane & 15;
    const int hi     = lane >> 4;
    const int waveM  = wave >> 2;              // 0..1
    const int waveN  = wave & 3;               // 0..3
    const int blockM = blockIdx.y * 128;
    const int blockN = blockIdx.x * 128;

    const int srow = t >> 1;
    const int sseg = (t & 1) << 4;             // 0 or 16

    v8f acc[4][2] = {};

    const int nch = n0 + n1;

    auto stage = [&](int c, int buf) {
        const float *Ap, *Wp;
        int lda, ldw, k0;
        if (c < n0) { Ap = A0; Wp = W0; lda = lda0; ldw = ldw0; k0 = c * 32; }
        else        { Ap = A1; Wp = W1; lda = HID;  ldw = HID;  k0 = (c - n0) * 32; }
        const float4* qa = (const float4*)(Ap + (size_t)(blockM + srow) * lda + k0 + sseg);
        const float4* qb = (const float4*)(Wp + (size_t)(blockN + srow) * ldw + k0 + sseg);
        float4 a0 = qa[0], a1 = qa[1], a2 = qa[2], a3 = qa[3];
        float4 b0 = qb[0], b1 = qb[1], b2 = qb[2], b3 = qb[3];
        uint4 pa0 = { pack2(a0.x, a0.y), pack2(a0.z, a0.w),
                      pack2(a1.x, a1.y), pack2(a1.z, a1.w) };
        uint4 pa1 = { pack2(a2.x, a2.y), pack2(a2.z, a2.w),
                      pack2(a3.x, a3.y), pack2(a3.z, a3.w) };
        uint4 pb0 = { pack2(b0.x, b0.y), pack2(b0.z, b0.w),
                      pack2(b1.x, b1.y), pack2(b1.z, b1.w) };
        uint4 pb1 = { pack2(b2.x, b2.y), pack2(b2.z, b2.w),
                      pack2(b3.x, b3.y), pack2(b3.z, b3.w) };
        *(uint4*)&As[buf][srow][sseg]     = pa0;   // ds_store_b128
        *(uint4*)&As[buf][srow][sseg + 8] = pa1;
        *(uint4*)&Bs[buf][srow][sseg]     = pb0;
        *(uint4*)&Bs[buf][srow][sseg + 8] = pb1;
    };

    stage(0, 0);
    __syncthreads();

    for (int c = 0; c < nch; ++c) {
        const int cur = c & 1;
        if (c + 1 < nch) stage(c + 1, cur ^ 1);   // overlap next-chunk loads with WMMA

        // A fragment: lane m16 = row M; halves at K = 8*hi..+7 and 16+8*hi..+7
        v16bf af[4];
        #pragma unroll
        for (int tm = 0; tm < 4; ++tm) {
            int r = waveM * 64 + tm * 16 + m16;
            v8bf x0 = *(const v8bf*)&As[cur][r][8 * hi];
            v8bf x1 = *(const v8bf*)&As[cur][r][16 + 8 * hi];
            af[tm] = __builtin_shufflevector(x0, x1,
                     0,1,2,3,4,5,6,7,8,9,10,11,12,13,14,15);
        }
        // B fragment: lane m16 = column N (weight row); halves at K = 8*hi..+7 and 16+8*hi..+7
        v16bf bfv[2];
        #pragma unroll
        for (int tn = 0; tn < 2; ++tn) {
            int r = waveN * 32 + tn * 16 + m16;
            v8bf y0 = *(const v8bf*)&Bs[cur][r][8 * hi];
            v8bf y1 = *(const v8bf*)&Bs[cur][r][16 + 8 * hi];
            bfv[tn] = __builtin_shufflevector(y0, y1,
                      0,1,2,3,4,5,6,7,8,9,10,11,12,13,14,15);
        }

        #pragma unroll
        for (int tm = 0; tm < 4; ++tm)
            #pragma unroll
            for (int tn = 0; tn < 2; ++tn)
                acc[tm][tn] = __builtin_amdgcn_wmma_f32_16x16x32_bf16(
                    false, af[tm], false, bfv[tn],
                    (short)0, acc[tm][tn], false, false);

        __syncthreads();
    }

    // D layout: VGPR v -> M = v + 8*hi, N = m16
    #pragma unroll
    for (int tm = 0; tm < 4; ++tm) {
        #pragma unroll
        for (int tn = 0; tn < 2; ++tn) {
            int col = blockN + waveN * 32 + tn * 16 + m16;
            #pragma unroll
            for (int v = 0; v < 8; ++v) {
                int rowg = blockM + waveM * 64 + tm * 16 + v + 8 * hi;
                gates[(size_t)rowg * GATES + col] = acc[tm][tn][v];
            }
        }
    }
}

// ---------------------------------------------------------------------------
// Kernel 3: LSTM gate nonlinearity + cell/hidden update (biases folded here)
// ---------------------------------------------------------------------------
__global__ void lstm_pointwise(const float* __restrict__ gates,
                               const float* __restrict__ c_in,
                               const float* __restrict__ bi,
                               const float* __restrict__ bh,
                               float* __restrict__ h_out,
                               float* __restrict__ c_out) {
    int i = blockIdx.x * blockDim.x + threadIdx.x;
    if (i >= BATCH * HID) return;
    int b = i >> 10, h = i & (HID - 1);
    const float* g = gates + (size_t)b * GATES;
    float gi = g[h]           + bi[h]           + bh[h];
    float gf = g[HID + h]     + bi[HID + h]     + bh[HID + h];
    float gg = g[2*HID + h]   + bi[2*HID + h]   + bh[2*HID + h];
    float go = g[3*HID + h]   + bi[3*HID + h]   + bh[3*HID + h];
    float ig = 1.f / (1.f + __expf(-gi));
    float fg = 1.f / (1.f + __expf(-gf));
    float og = 1.f / (1.f + __expf(-go));
    float c  = fg * c_in[i] + ig * tanhf(gg);
    c_out[i] = c;
    h_out[i] = og * tanhf(c);
}

// ---------------------------------------------------------------------------
// Kernel 4: mu / sigma head over feature-major, layer-minor flatten
// ---------------------------------------------------------------------------
__global__ void head_kernel(const float* __restrict__ hstack,
                            const float* __restrict__ w_mu, const float* __restrict__ b_mu,
                            const float* __restrict__ w_ps, const float* __restrict__ b_ps,
                            float* __restrict__ mu, float* __restrict__ sigma) {
    __shared__ float smu[256];
    __shared__ float sps[256];
    const int b = blockIdx.x, t = threadIdx.x;
    float am = 0.f, ap = 0.f;
    for (int h = t; h < HID; h += 256) {
        #pragma unroll
        for (int l = 0; l < LAYERS; ++l) {
            float v = hstack[(size_t)l * (BATCH * HID) + (size_t)b * HID + h];
            am += v * w_mu[h * LAYERS + l];
            ap += v * w_ps[h * LAYERS + l];
        }
    }
    smu[t] = am; sps[t] = ap;
    __syncthreads();
    for (int s = 128; s > 0; s >>= 1) {
        if (t < s) { smu[t] += smu[t + s]; sps[t] += sps[t + s]; }
        __syncthreads();
    }
    if (t == 0) {
        mu[b] = smu[0] + b_mu[0];
        float v = sps[0] + b_ps[0];
        sigma[b] = (v > 0.f) ? (v + log1pf(__expf(-v))) : log1pf(__expf(v));
    }
}

// ---------------------------------------------------------------------------
extern "C" void kernel_launch(void* const* d_in, const int* in_sizes, int n_in,
                              void* d_out, int out_size, void* d_ws, size_t ws_size,
                              hipStream_t stream) {
    (void)in_sizes; (void)n_in; (void)out_size; (void)ws_size;
    const float* x      = (const float*)d_in[0];
    const int*   idx    = (const int*)d_in[1];
    const float* hidden = (const float*)d_in[2];
    const float* cell   = (const float*)d_in[3];
    const float* emb    = (const float*)d_in[4];
    const float* w_ih0  = (const float*)d_in[5];
    const float* w_hh0  = (const float*)d_in[6];
    const float* w_ihr  = (const float*)d_in[7];
    const float* w_hhr  = (const float*)d_in[8];
    const float* b_ih   = (const float*)d_in[9];
    const float* b_hh   = (const float*)d_in[10];
    const float* w_mu   = (const float*)d_in[11];
    const float* b_mu   = (const float*)d_in[12];
    const float* w_ps   = (const float*)d_in[13];
    const float* b_ps   = (const float*)d_in[14];

    float* out   = (float*)d_out;
    float* mu    = out;
    float* sigma = out + BATCH;
    float* hstk  = out + 2 * BATCH;                         // [3,1024,1024]
    float* cstk  = hstk + (size_t)LAYERS * BATCH * HID;     // [3,1024,1024]

    float* inp0  = (float*)d_ws;                            // 1024 x 96
    float* w0p   = inp0 + BATCH * IN_PAD;                   // 4096 x 96
    float* gates = w0p + (size_t)GATES * IN_PAD;            // 1024 x 4096

    build_input<<<(BATCH * IN_PAD + 255) / 256, 256, 0, stream>>>(x, idx, emb, inp0);
    pad_w0<<<(GATES * IN_PAD + 255) / 256, 256, 0, stream>>>(w_ih0, w0p);

    for (int l = 0; l < LAYERS; ++l) {
        const float* A0 = (l == 0) ? inp0 : (hstk + (size_t)(l - 1) * BATCH * HID);
        int lda0        = (l == 0) ? IN_PAD : HID;
        int n0          = (l == 0) ? (IN_PAD / 32) : (HID / 32);
        const float* W0 = (l == 0) ? w0p : (w_ihr + (size_t)(l - 1) * GATES * HID);
        int ldw0        = (l == 0) ? IN_PAD : HID;
        const float* A1 = hidden + (size_t)l * BATCH * HID;
        const float* W1 = (l == 0) ? w_hh0 : (w_hhr + (size_t)(l - 1) * GATES * HID);

        gates_gemm<<<dim3(GATES / 128, BATCH / 128), 256, 0, stream>>>(
            A0, lda0, n0, W0, ldw0, A1, W1, HID / 32, gates);

        lstm_pointwise<<<(BATCH * HID + 255) / 256, 256, 0, stream>>>(
            gates, cell + (size_t)l * BATCH * HID,
            b_ih + (size_t)l * GATES, b_hh + (size_t)l * GATES,
            hstk + (size_t)l * BATCH * HID, cstk + (size_t)l * BATCH * HID);
    }

    head_kernel<<<BATCH, 256, 0, stream>>>(hstk, w_mu, b_mu, w_ps, b_ps, mu, sigma);
}
